// My_MLP_69320772157909
// MI455X (gfx1250) — matrix operand order repaired and checked by reading, not compiled, not run on hardware
//
#include <hip/hip_runtime.h>

typedef __attribute__((ext_vector_type(16))) _Float16 v16h;
typedef __attribute__((ext_vector_type(8)))  _Float16 v8h;
typedef __attribute__((ext_vector_type(8)))  float    v8f;

#define N_NODES 8192
#define DIM     512
#define SEG_SHIFT 11      // 2048 nodes per client subgraph
#define TOPK    31        // k + 1
#define INF     __builtin_inff()

#define LDS_STRIDE_H 72   // 64 halves of K-window + 8 halves pad (144 B rows)

// ---------------------------------------------------------------------------
// CDNA5 async-copy helper, GVS form: SGPR 64-bit base + 32-bit VGPR offset.
// Keeps uniform address math on the SALU; one v_add_nc_u32 per stage advance.
// ---------------------------------------------------------------------------
__device__ __forceinline__ void async_load_b128_s(uint32_t lds_addr,
                                                  uint32_t voff,
                                                  uint64_t sbase) {
  asm volatile("global_load_async_to_lds_b128 %0, %1, %2"
               :: "v"(lds_addr), "v"(voff), "s"(sbase) : "memory");
}
__device__ __forceinline__ uint32_t lds_addr_of(const void* p) {
  // flat shared-aperture pointers carry the LDS byte offset in addr[31:0]
  return (uint32_t)(uintptr_t)p;
}

// ---------------------------------------------------------------------------
// Kernel 1: h = relu(f*w0)*w1 ; emb = h / ||h||_2  -> f16 for WMMA
// One wave32 per row (512 elems -> 16 per lane), wave-reduction via shfl_xor.
// ---------------------------------------------------------------------------
__global__ __launch_bounds__(256) void prep_emb(const float* __restrict__ feat,
                                                const float* __restrict__ w0,
                                                const float* __restrict__ w1,
                                                _Float16* __restrict__ emb) {
  const int wave = threadIdx.x >> 5;
  const int lane = threadIdx.x & 31;
  const int row  = blockIdx.x * 8 + wave;

  const float* fr = feat + (size_t)row * DIM;
  float h[16];
  float acc = 0.f;
#pragma unroll
  for (int i = 0; i < 16; ++i) {
    const int idx = i * 32 + lane;
    float x = fr[idx] * w0[idx];
    x = x > 0.f ? x : 0.f;
    x *= w1[idx];
    h[i] = x;
    acc += x * x;
  }
#pragma unroll
  for (int off = 16; off > 0; off >>= 1) acc += __shfl_xor(acc, off, 32);
  const float s = rsqrtf(acc + 1e-24f);

  _Float16* er = emb + (size_t)row * DIM;
#pragma unroll
  for (int i = 0; i < 16; ++i) er[i * 32 + lane] = (_Float16)(h[i] * s);
}

// ---------------------------------------------------------------------------
// Kernel 2: sim = emb @ emb^T via v_wmma_f32_16x16x32_f16.
//
// Block = 256 threads (8 waves) computes a 128x128 output tile; K staged
// 64-wide through LDS with double-buffered global_load_async_to_lds_b128
// (GVS addressing, ASYNCcnt pipeline, s_wait_asynccnt 8/0).  Wave w owns
// M-rows [16w,16w+16) x all 128 N cols: A fragment reused across 8 N-tiles,
// B tile shared by all 8 waves via LDS.
//
// Fragment layouts per CDNA5 ISA 7.12.2 (wave32):
//   A 16x32 f16: lanes 0-15 -> M=lane, K {0..7,16..23}; lanes 16-31 K {8..15,24..31}
//   B 32x16 f16: lanes 0-15 -> N=lane, K {0..7,16..23}; lanes 16-31 K {8..15,24..31}
// LDS rows padded to 144 B so 16-lane b128 fragment reads are bank-conflict-free.
// ---------------------------------------------------------------------------
__global__ __launch_bounds__(256) void gemm_sim(const _Float16* __restrict__ emb,
                                                float* __restrict__ out) {
  __shared__ __align__(16) _Float16 sA[2][128 * LDS_STRIDE_H];
  __shared__ __align__(16) _Float16 sB[2][128 * LDS_STRIDE_H];

  const int tid  = threadIdx.x;
  const int lane = tid & 31;
  const int wave = tid >> 5;
  const int l15  = lane & 15;
  const int hi   = lane >> 4;
  const int rowBase = blockIdx.y * 128;
  const int colBase = blockIdx.x * 128;

  // Uniform tile bases -> SGPR pairs (GVS saddr operand).
  const uint64_t baseA = (uint64_t)(uintptr_t)emb + (uint64_t)rowBase * (DIM * 2);
  const uint64_t baseB = (uint64_t)(uintptr_t)emb + (uint64_t)colBase * (DIM * 2);

  // Per-thread DMA chunk descriptors: 1024 x 16B chunks per tile stage,
  // 4 per thread; chunk c -> tile row c>>3, 16B sub-chunk c&7.
  uint32_t vOff[4];              // byte offset within tile (32-bit, per thread)
  uint32_t lA[2][4], lB[2][4];   // LDS destinations for both buffers
#pragma unroll
  for (int i = 0; i < 4; ++i) {
    const int c   = tid + 256 * i;
    const int row = c >> 3;
    const int off = c & 7;                               // 8 halves per chunk
    vOff[i] = (uint32_t)row * (DIM * 2) + (uint32_t)off * 16;
    const uint32_t l = (uint32_t)row * (LDS_STRIDE_H * 2) + off * 16;
    lA[0][i] = lds_addr_of(&sA[0][0]) + l;
    lA[1][i] = lds_addr_of(&sA[1][0]) + l;
    lB[0][i] = lds_addr_of(&sB[0][0]) + l;
    lB[1][i] = lds_addr_of(&sB[1][0]) + l;
  }

  auto issue_stage = [&](int buf, int kk) {
    const uint32_t koff = (uint32_t)kk * 2;              // byte advance along K
#pragma unroll
    for (int i = 0; i < 4; ++i) {
      async_load_b128_s(lA[buf][i], vOff[i] + koff, baseA);
      async_load_b128_s(lB[buf][i], vOff[i] + koff, baseB);
    }
  };

  v8f acc[8];
  const v8f vzero = {};
#pragma unroll
  for (int n = 0; n < 8; ++n) acc[n] = vzero;

  issue_stage(0, 0);                                     // prologue
  int buf = 0;
  for (int kk = 0; kk < DIM; kk += 64) {
    if (kk + 64 < DIM) {
      issue_stage(buf ^ 1, kk + 64);
      asm volatile("s_wait_asynccnt 0x8" ::: "memory");  // current stage landed
    } else {
      asm volatile("s_wait_asynccnt 0x0" ::: "memory");
    }
    __syncthreads();

    const _Float16* A = sA[buf];
    const _Float16* B = sB[buf];
#pragma unroll
    for (int kh = 0; kh < 2; ++kh) {                     // two k=32 sub-steps
      const int abase = (wave * 16 + l15) * LDS_STRIDE_H + kh * 32 + hi * 8;
      v8h a0 = *(const v8h*)(A + abase);
      v8h a1 = *(const v8h*)(A + abase + 16);
      v16h a;
#pragma unroll
      for (int i = 0; i < 8; ++i) { a[i] = a0[i]; a[8 + i] = a1[i]; }

#pragma unroll
      for (int n = 0; n < 8; ++n) {
        const int bbase = (n * 16 + l15) * LDS_STRIDE_H + kh * 32 + hi * 8;
        v8h b0 = *(const v8h*)(B + bbase);
        v8h b1 = *(const v8h*)(B + bbase + 16);
        v16h b;
#pragma unroll
        for (int i = 0; i < 8; ++i) { b[i] = b0[i]; b[8 + i] = b1[i]; }
        acc[n] = __builtin_amdgcn_wmma_f32_16x16x32_f16(
            false, a, false, b, (short)0, acc[n], false, false);
      }
    }
    __syncthreads();                                     // buffer safe to refill
    buf ^= 1;
  }

  // C/D layout: VGPR v -> M = v + (lane<16 ? 0 : 8), N = lane&15.
  // NT stores: 256 MB sim stream must not evict the L2-resident emb.
#pragma unroll
  for (int n = 0; n < 8; ++n) {
    float* orow = out + (size_t)(rowBase + wave * 16 + hi * 8) * N_NODES
                      + (colBase + n * 16 + l15);
#pragma unroll
    for (int v = 0; v < 8; ++v)
      __builtin_nontemporal_store(acc[n][v], orow + (size_t)v * N_NODES);
  }
}

// ---------------------------------------------------------------------------
// Kernel 3: per-row cross-segment top-31 threshold + mask + relu, in place.
// One block (8 waves) per row; 32 column values per thread held in VGPRs;
// 31 rounds of predicated max-extraction (register scan + shfl + 8-slot LDS).
// Final: out = (cross && v >= thresh && v > 0) ? v : 0.
// ---------------------------------------------------------------------------
__global__ __launch_bounds__(256) void topk_relu(float* __restrict__ out) {
  const int row  = blockIdx.x;
  const int tid  = threadIdx.x;
  const int lane = tid & 31;
  const int wave = tid >> 5;
  const int seg  = row >> SEG_SHIFT;

  float* r = out + (size_t)row * N_NODES;

  float v[32], key[32];
#pragma unroll
  for (int i = 0; i < 32; ++i) {
    const int col = i * 256 + tid;                // coalesced
    const float x = __builtin_nontemporal_load(r + col);
    v[i]   = x;
    key[i] = ((col >> SEG_SHIFT) != seg) ? x : -INF;
  }

  __shared__ float red[8];
  float cur = INF;
  for (int it = 0; it < TOPK; ++it) {
    float m = -INF;
#pragma unroll
    for (int i = 0; i < 32; ++i)
      if (key[i] < cur) m = fmaxf(m, key[i]);
#pragma unroll
    for (int off = 16; off > 0; off >>= 1) m = fmaxf(m, __shfl_xor(m, off, 32));
    if (lane == 0) red[wave] = m;
    __syncthreads();
    if (tid == 0) {
      float x = red[0];
      for (int k = 1; k < 8; ++k) x = fmaxf(x, red[k]);
      red[0] = x;
    }
    __syncthreads();
    cur = red[0];
    __syncthreads();
  }

#pragma unroll
  for (int i = 0; i < 32; ++i) {
    const int col = i * 256 + tid;
    const float x = v[i];
    const bool keep = (key[i] > -INF) && (x >= cur) && (x > 0.f);
    __builtin_nontemporal_store(keep ? x : 0.f, r + col);
  }
}

// ---------------------------------------------------------------------------
extern "C" void kernel_launch(void* const* d_in, const int* in_sizes, int n_in,
                              void* d_out, int out_size, void* d_ws, size_t ws_size,
                              hipStream_t stream) {
  const float* feat = (const float*)d_in[0];   // [8192, 512] f32
  const float* w0   = (const float*)d_in[1];   // [512] f32
  const float* w1   = (const float*)d_in[2];   // [512] f32
  float*       out  = (float*)d_out;           // [8192, 8192] f32
  _Float16*    emb  = (_Float16*)d_ws;         // [8192, 512] f16 scratch (8 MB)

  prep_emb<<<N_NODES / 8, 256, 0, stream>>>(feat, w0, w1, emb);

  dim3 g(N_NODES / 128, N_NODES / 128);        // 64 x 64 blocks, 128x128 tiles
  gemm_sim<<<g, 256, 0, stream>>>(emb, out);

  topk_relu<<<N_NODES, 256, 0, stream>>>(out);
}
